// BlockDecoder_3547642987208
// MI455X (gfx1250) — matrix operand order repaired and checked by reading, not compiled, not run on hardware
//
#include <hip/hip_runtime.h>
#include <math.h>

// ---------------------------------------------------------------------------
// Types for gfx1250 WMMA (wave32).
// ---------------------------------------------------------------------------
typedef __attribute__((ext_vector_type(16))) __bf16          bf16x16;
typedef __attribute__((ext_vector_type(16))) unsigned short  u16x16;
typedef __attribute__((ext_vector_type(8)))  float           f32x8;

#define WMMA_BF16(a, b, c)                                                     \
  __builtin_amdgcn_wmma_f32_16x16x32_bf16(                                     \
      false, __builtin_bit_cast(bf16x16, (a)),                                 \
      false, __builtin_bit_cast(bf16x16, (b)),                                 \
      (short)0, (c), false, false)

__device__ __forceinline__ unsigned short f2bf(float f) {
  unsigned int u = __float_as_uint(f);
  u += 0x7FFFu + ((u >> 16) & 1u);      // round-to-nearest-even
  return (unsigned short)(u >> 16);
}

// ---------------------------------------------------------------------------
// fp32 -> bf16 conversion (activations staged once in bf16).
// ---------------------------------------------------------------------------
__global__ void cvt_f32_bf16(const float* __restrict__ in,
                             unsigned short* __restrict__ out, int n) {
  int i = blockIdx.x * blockDim.x + threadIdx.x;
  int stride = gridDim.x * blockDim.x;
  for (; i < n; i += stride) out[i] = f2bf(in[i]);
}

// ---------------------------------------------------------------------------
// Generic bf16 WMMA GEMM: C[M,N] = A[M,K](bf16) * B[K,N](f32 weights,
// converted to bf16 while staging in LDS). 128x128 block tile, 8 waves
// (2x4), each wave 64x32 via 4x2 16x16 WMMA accumulators, K-step 32.
// A staged [m][k] with b128 loads; B staged TRANSPOSED ([n][k]) from
// float4 (global_load_b128) coalesced reads so every WMMA fragment gather
// is K-contiguous and 16B-aligned -> ds_load_b128.
// Epilogue fuses bias + residual + relu and writes f32 and/or bf16.
// bPerHead: B stored as [H=NumCols/64, K, 64] (stacked per-head qkv
// weights), else row-major [K, N].
// ---------------------------------------------------------------------------
__global__ __launch_bounds__(256) void gemm_bf16_wmma(
    const unsigned short* __restrict__ A, const float* __restrict__ B,
    int M, int N, int K, int bPerHead,
    const float* __restrict__ bias, const float* __restrict__ residual,
    int relu, float* __restrict__ outF, unsigned short* __restrict__ outB) {
  __shared__ unsigned short As[128][48];   // [m][k], 96B row stride (16B mult)
  __shared__ unsigned short Bs[128][48];   // [n][k] (transposed)

  const int tid  = threadIdx.x;
  const int lane = tid & 31, wid = tid >> 5;
  const int wm = wid >> 2, wn = wid & 3;   // 2x4 wave grid
  const int lh = lane & 15, hi = lane >> 4;
  const int row0 = blockIdx.y * 128, n0 = blockIdx.x * 128;

  f32x8 acc[4][2];
#pragma unroll
  for (int i = 0; i < 4; ++i)
#pragma unroll
    for (int j = 0; j < 2; ++j)
#pragma unroll
      for (int e = 0; e < 8; ++e) acc[i][j][e] = 0.0f;

  for (int k0 = 0; k0 < K; k0 += 32) {
    __syncthreads();
    // Stage A: 128x32 bf16, 8-elem (16B) vector chunks.
#pragma unroll
    for (int c = 0; c < 2; ++c) {
      int g = tid + c * 256;               // 512 groups of 8 bf16
      int r = g >> 2;                      // 4 groups per row
      int c8 = (g & 3) << 3;
      *(uint4*)&As[r][c8] =
          *(const uint4*)&A[(size_t)(row0 + r) * K + (k0 + c8)];
    }
    // Stage B: float4 coalesced global reads, transposed scalar LDS stores.
#pragma unroll
    for (int c = 0; c < 4; ++c) {
      int g = tid + c * 256;               // 1024 groups of 4 f32
      int kk = g >> 5;                     // 32 groups per k-row
      int n4 = (g & 31) << 2;
      size_t col = (size_t)(n0 + n4);
      size_t idx = bPerHead
                       ? ((col >> 6) * (size_t)K + (size_t)(k0 + kk)) * 64 +
                             (col & 63)
                       : (size_t)(k0 + kk) * (size_t)N + col;
      float4 bv = *(const float4*)&B[idx];
      Bs[n4 + 0][kk] = f2bf(bv.x);
      Bs[n4 + 1][kk] = f2bf(bv.y);
      Bs[n4 + 2][kk] = f2bf(bv.z);
      Bs[n4 + 3][kk] = f2bf(bv.w);
    }
    __syncthreads();

    // Fragments per ISA 7.12.2 layouts; all gathers K-contiguous -> b128.
    u16x16 af[4], bf[2];
#pragma unroll
    for (int i = 0; i < 4; ++i)
#pragma unroll
      for (int e = 0; e < 16; ++e) {
        int k = hi * 8 + (e < 8 ? e : e + 8);
        af[i][e] = As[wm * 64 + i * 16 + lh][k];
      }
#pragma unroll
    for (int j = 0; j < 2; ++j)
#pragma unroll
      for (int e = 0; e < 16; ++e)
        bf[j][e] = Bs[wn * 32 + j * 16 + lh][hi * 8 + (e < 8 ? e : e + 8)];
#pragma unroll
    for (int i = 0; i < 4; ++i)
#pragma unroll
      for (int j = 0; j < 2; ++j) acc[i][j] = WMMA_BF16(af[i], bf[j], acc[i][j]);
  }

  // Epilogue.
#pragma unroll
  for (int i = 0; i < 4; ++i)
#pragma unroll
    for (int j = 0; j < 2; ++j)
#pragma unroll
      for (int r = 0; r < 8; ++r) {
        int m = wm * 64 + i * 16 + r + hi * 8;
        int n = wn * 32 + j * 16 + lh;
        size_t row = (size_t)(row0 + m), col = (size_t)(n0 + n);
        float v = acc[i][j][r];
        if (bias) v += bias[col];
        if (residual) v += residual[row * N + col];
        if (relu) v = fmaxf(v, 0.0f);
        if (outF) outF[row * N + col] = v;
        if (outB) outB[row * N + col] = f2bf(v);
      }
}

// ---------------------------------------------------------------------------
// Fused flash attention (bf16 WMMA, f32 online softmax, wave32).
// Q,K,V,O: bf16 [B, S(or T), H*HD] row-major. One block = (64 Q rows, h, b),
// 4 waves, each wave owns 16 Q rows. Streams 64-wide K/V tiles through LDS.
// V is staged transposed (Vst[d][t]) so P.V B-fragments are K-contiguous.
// 160B LDS row stride keeps all fragment ds_load_b128 16B-aligned.
// Scores are scaled by sqrt(HD)=8 as in the reference.
// ---------------------------------------------------------------------------
template <bool CAUSAL>
__global__ __launch_bounds__(128) void attn_kernel(
    const unsigned short* __restrict__ Q, const unsigned short* __restrict__ K,
    const unsigned short* __restrict__ V, unsigned short* __restrict__ O,
    int S, int Tkv) {
  const int E = 1024, HD = 64;
  const int s0 = blockIdx.x * 64;
  const int h = blockIdx.y;
  const int b = blockIdx.z;
  const size_t qbase = ((size_t)b * S) * E + (size_t)h * HD;
  const size_t kbase = ((size_t)b * Tkv) * E + (size_t)h * HD;

  __shared__ unsigned short Qs[64][80];      // [s][d]
  __shared__ unsigned short Ks[64][80];      // [t][d]
  __shared__ unsigned short Vst[64][80];     // [d][t]  (transposed)
  __shared__ unsigned short Ps[4][16][80];   // per-wave P strip [m][t]

  const int tid = threadIdx.x;               // 128 threads, 4 waves
  const int lane = tid & 31, w = tid >> 5;
  const int lh = lane & 15, hi = lane >> 4;

  // Load Q tile (64x64 bf16) with b128 chunks.
#pragma unroll
  for (int c = 0; c < 4; ++c) {
    int g = tid + c * 128;                   // 512 groups of 8
    int r = g >> 3;
    int c8 = (g & 7) << 3;
    *(uint4*)&Qs[r][c8] = *(const uint4*)&Q[qbase + (size_t)(s0 + r) * E + c8];
  }
  __syncthreads();

  // Q A-fragments (2 K-chunks of 32 over HD=64), loaded once.
  u16x16 aq[2];
#pragma unroll
  for (int kk = 0; kk < 2; ++kk)
#pragma unroll
    for (int e = 0; e < 16; ++e) {
      int d = kk * 32 + hi * 8 + (e < 8 ? e : e + 8);
      aq[kk][e] = Qs[w * 16 + lh][d];
    }

  float mi[8], li[8];
  f32x8 accO[4];
#pragma unroll
  for (int r = 0; r < 8; ++r) { mi[r] = -3.0e38f; li[r] = 0.0f; }
#pragma unroll
  for (int jd = 0; jd < 4; ++jd)
#pragma unroll
    for (int e = 0; e < 8; ++e) accO[jd][e] = 0.0f;

  const int tEnd = CAUSAL ? (s0 + 64) : Tkv;
  for (int t0 = 0; t0 < tEnd; t0 += 64) {
    __syncthreads();
#pragma unroll
    for (int c = 0; c < 4; ++c) {            // stage K ([t][d]) and V ([d][t])
      int g = tid + c * 128;
      int r = g >> 3;
      int c8 = (g & 7) << 3;
      *(uint4*)&Ks[r][c8] = *(const uint4*)&K[kbase + (size_t)(t0 + r) * E + c8];
      uint4 vv = *(const uint4*)&V[kbase + (size_t)(t0 + r) * E + c8];
      const unsigned short* vp = (const unsigned short*)&vv;
#pragma unroll
      for (int i = 0; i < 8; ++i) Vst[c8 + i][r] = vp[i];
    }
    __syncthreads();

    // Scores: S_tile(16x64) = Q(16x64) @ K^T (transpose via LDS indexing;
    // gather is d-contiguous -> b128).
    f32x8 sc[4];
#pragma unroll
    for (int j = 0; j < 4; ++j) {
#pragma unroll
      for (int e = 0; e < 8; ++e) sc[j][e] = 0.0f;
#pragma unroll
      for (int kk = 0; kk < 2; ++kk) {
        u16x16 bu;
#pragma unroll
        for (int e = 0; e < 16; ++e)
          bu[e] = Ks[j * 16 + lh][kk * 32 + hi * 8 + (e < 8 ? e : e + 8)];
        sc[j] = WMMA_BF16(aq[kk], bu, sc[j]);
      }
    }

    // Online softmax: each C row lives in one VGPR across one 16-lane half.
#pragma unroll
    for (int r = 0; r < 8; ++r) {
      int srow = s0 + w * 16 + r + hi * 8;
      float vals[4];
      float rmax = -3.0e38f;
#pragma unroll
      for (int j = 0; j < 4; ++j) {
        float v = sc[j][r] * 8.0f;           // * sqrt(HD)
        if (CAUSAL) {
          int tcol = t0 + j * 16 + lh;
          if (tcol > srow) v = -3.0e38f;
        }
        vals[j] = v;
        rmax = fmaxf(rmax, v);
      }
#pragma unroll
      for (int off = 8; off >= 1; off >>= 1)
        rmax = fmaxf(rmax, __shfl_xor(rmax, off, 32));
      float mnew = fmaxf(mi[r], rmax);
      float rsum = 0.0f;
#pragma unroll
      for (int j = 0; j < 4; ++j) {
        float p = __expf(vals[j] - mnew);
        Ps[w][r + hi * 8][j * 16 + lh] = f2bf(p);
        rsum += p;
      }
#pragma unroll
      for (int off = 8; off >= 1; off >>= 1) rsum += __shfl_xor(rsum, off, 32);
      float corr = __expf(mi[r] - mnew);
      li[r] = li[r] * corr + rsum;
#pragma unroll
      for (int jd = 0; jd < 4; ++jd) accO[jd][r] *= corr;
      mi[r] = mnew;
    }
    __syncthreads();                          // P visible before PV gather

    // O += P(16x64) @ V(64x64); both gathers K(t)-contiguous -> b128.
#pragma unroll
    for (int kk = 0; kk < 2; ++kk) {
      u16x16 ap;
#pragma unroll
      for (int e = 0; e < 16; ++e) {
        int t = kk * 32 + hi * 8 + (e < 8 ? e : e + 8);
        ap[e] = Ps[w][lh][t];
      }
#pragma unroll
      for (int jd = 0; jd < 4; ++jd) {
        u16x16 bv;
#pragma unroll
        for (int e = 0; e < 16; ++e)
          bv[e] = Vst[jd * 16 + lh][kk * 32 + hi * 8 + (e < 8 ? e : e + 8)];
        accO[jd] = WMMA_BF16(ap, bv, accO[jd]);
      }
    }
  }

  // Normalize and write O (bf16).
#pragma unroll
  for (int jd = 0; jd < 4; ++jd)
#pragma unroll
    for (int r = 0; r < 8; ++r) {
      int m = w * 16 + r + hi * 8;
      float v = accO[jd][r] / li[r];
      O[qbase + (size_t)(s0 + m) * E + jd * 16 + lh] = f2bf(v);
    }
}

// ---------------------------------------------------------------------------
// Row LayerNorm over E=1024; writes f32 and optionally bf16.
// ---------------------------------------------------------------------------
__global__ __launch_bounds__(256) void ln_kernel(
    const float* __restrict__ in, const float* __restrict__ g,
    const float* __restrict__ bt, float* __restrict__ outF,
    unsigned short* __restrict__ outB) {
  const int E = 1024;
  const size_t row = blockIdx.x;
  const float* xr = in + row * E;
  float s = 0.0f, s2 = 0.0f;
  for (int i = threadIdx.x; i < E; i += 256) {
    float v = xr[i];
    s += v;
    s2 += v * v;
  }
  __shared__ float rs[256], rs2[256];
  rs[threadIdx.x] = s;
  rs2[threadIdx.x] = s2;
  __syncthreads();
  for (int o = 128; o > 0; o >>= 1) {
    if (threadIdx.x < o) {
      rs[threadIdx.x] += rs[threadIdx.x + o];
      rs2[threadIdx.x] += rs2[threadIdx.x + o];
    }
    __syncthreads();
  }
  float mean = rs[0] * (1.0f / E);
  float var = rs2[0] * (1.0f / E) - mean * mean;
  float inv = rsqrtf(var + 1e-5f);
  for (int i = threadIdx.x; i < E; i += 256) {
    float v = (xr[i] - mean) * inv * g[i] + bt[i];
    if (outF) outF[row * E + i] = v;
    if (outB) outB[row * E + i] = f2bf(v);
  }
}

// ---------------------------------------------------------------------------
// Driver. Workspace layout (needs 128 MB):
//   [0,8)    xb    bf16 current attention input
//   [8,16)   zb    bf16 encoder memory
//   [16,24)  Qb    [24,32) Kb   [32,40) Vb   [40,48) Ob   (bf16)
//   [48,64)  attnF f32 (proj+residual, reused for FFN output)
//   [64,80)  x1F   f32
//   [80,96)  x2F   f32
//   [96,128) hidden bf16 [4096,4096]
// ---------------------------------------------------------------------------
extern "C" void kernel_launch(void* const* d_in, const int* in_sizes, int n_in,
                              void* d_out, int out_size, void* d_ws,
                              size_t ws_size, hipStream_t stream) {
  const int S = 1024, E = 1024;
  const int M = 4 * S;                       // B*S = B*T = 4096

  const float* x      = (const float*)d_in[0];
  const float* z      = (const float*)d_in[1];
  const float* mha_wq = (const float*)d_in[2];
  const float* mha_wk = (const float*)d_in[3];
  const float* mha_wv = (const float*)d_in[4];
  const float* mha_pw = (const float*)d_in[5];
  const float* mha_pb = (const float*)d_in[6];
  const float* ca_wq  = (const float*)d_in[7];
  const float* ca_wk  = (const float*)d_in[8];
  const float* ca_wv  = (const float*)d_in[9];
  const float* ca_pw  = (const float*)d_in[10];
  const float* ca_pb  = (const float*)d_in[11];
  const float* fc1_w  = (const float*)d_in[12];
  const float* fc1_b  = (const float*)d_in[13];
  const float* fc2_w  = (const float*)d_in[14];
  const float* fc2_b  = (const float*)d_in[15];
  const float* ln1_g  = (const float*)d_in[16];
  const float* ln1_b  = (const float*)d_in[17];
  const float* ln2_g  = (const float*)d_in[18];
  const float* ln2_b  = (const float*)d_in[19];
  const float* ln3_g  = (const float*)d_in[20];
  const float* ln3_b  = (const float*)d_in[21];

  char* w = (char*)d_ws;
  const size_t MB = 1024ull * 1024ull;
  unsigned short* xb     = (unsigned short*)(w + 0 * MB);
  unsigned short* zb     = (unsigned short*)(w + 8 * MB);
  unsigned short* Qb     = (unsigned short*)(w + 16 * MB);
  unsigned short* Kb     = (unsigned short*)(w + 24 * MB);
  unsigned short* Vb     = (unsigned short*)(w + 32 * MB);
  unsigned short* Ob     = (unsigned short*)(w + 40 * MB);
  float*          attnF  = (float*)(w + 48 * MB);
  float*          x1F    = (float*)(w + 64 * MB);
  float*          x2F    = (float*)(w + 80 * MB);
  unsigned short* hidden = (unsigned short*)(w + 96 * MB);

  const dim3 blk256(256);
  const dim3 gN1k(1024 / 128, M / 128);      // N=1024 GEMMs
  const dim3 gN4k(4096 / 128, M / 128);      // N=4096 GEMM
  const dim3 gAttn(S / 64, 16, 4);

  // bf16 staging of activations.
  cvt_f32_bf16<<<4096, blk256, 0, stream>>>(x, xb, M * E);
  cvt_f32_bf16<<<4096, blk256, 0, stream>>>(z, zb, M * E);

  // ---- masked self-attention ----
  gemm_bf16_wmma<<<gN1k, blk256, 0, stream>>>(xb, mha_wq, M, 1024, 1024, 1,
                                              nullptr, nullptr, 0, nullptr, Qb);
  gemm_bf16_wmma<<<gN1k, blk256, 0, stream>>>(xb, mha_wk, M, 1024, 1024, 1,
                                              nullptr, nullptr, 0, nullptr, Kb);
  gemm_bf16_wmma<<<gN1k, blk256, 0, stream>>>(xb, mha_wv, M, 1024, 1024, 1,
                                              nullptr, nullptr, 0, nullptr, Vb);
  attn_kernel<true><<<gAttn, dim3(128), 0, stream>>>(Qb, Kb, Vb, Ob, S, S);
  gemm_bf16_wmma<<<gN1k, blk256, 0, stream>>>(Ob, mha_pw, M, 1024, 1024, 0,
                                              mha_pb, x, 0, attnF, nullptr);
  ln_kernel<<<M, blk256, 0, stream>>>(attnF, ln1_g, ln1_b, x1F, xb);

  // ---- cross-attention ----
  gemm_bf16_wmma<<<gN1k, blk256, 0, stream>>>(xb, ca_wq, M, 1024, 1024, 1,
                                              nullptr, nullptr, 0, nullptr, Qb);
  gemm_bf16_wmma<<<gN1k, blk256, 0, stream>>>(zb, ca_wk, M, 1024, 1024, 1,
                                              nullptr, nullptr, 0, nullptr, Kb);
  gemm_bf16_wmma<<<gN1k, blk256, 0, stream>>>(zb, ca_wv, M, 1024, 1024, 1,
                                              nullptr, nullptr, 0, nullptr, Vb);
  attn_kernel<false><<<gAttn, dim3(128), 0, stream>>>(Qb, Kb, Vb, Ob, S, S);
  gemm_bf16_wmma<<<gN1k, blk256, 0, stream>>>(Ob, ca_pw, M, 1024, 1024, 0,
                                              ca_pb, x1F, 0, attnF, nullptr);
  ln_kernel<<<M, blk256, 0, stream>>>(attnF, ln2_g, ln2_b, x2F, xb);

  // ---- feed-forward ----
  gemm_bf16_wmma<<<gN4k, blk256, 0, stream>>>(xb, fc1_w, M, 4096, 1024, 0,
                                              fc1_b, nullptr, 1, nullptr,
                                              hidden);
  gemm_bf16_wmma<<<gN1k, blk256, 0, stream>>>(hidden, fc2_w, M, 1024, 4096, 0,
                                              fc2_b, x2F, 0, attnF, nullptr);
  ln_kernel<<<M, blk256, 0, stream>>>(attnF, ln3_g, ln3_b, (float*)d_out,
                                      nullptr);
}
